// Net_75230647156957
// MI455X (gfx1250) — compile-verified, matrix operand order checked by reading, not run on hardware
//
#include <hip/hip_runtime.h>
#include <hip/hip_bf16.h>

// ---------------------------------------------------------------------------
// Fused MoE forward for MI455X (gfx1250, wave32, WMMA 16x16x32 f16->f32).
//
// Input order assumption (setup_inputs dict order): d_in[0] = x [131072,256],
// then params in insertion order (s_w1, s_b1, s_g1, s_be1, s_w2, ... , hyi_b2).
//
// Kernel 1 packs all weight matrices as f16 WMMA B-fragments (32x16 tiles,
// 512 halves/tile, lane-major) into d_ws.  Kernel 2: one wave = 16 rows,
// whole network fused; activations live in per-wave LDS as f16, LayerNorm
// staging in f32.  All matmuls use v_wmma_f32_16x16x32_f16.  B fragments are
// loaded into independent register arrays so the loads clause and overlap
// with WMMA issue; weight stages are prefetched ahead of use; the one-shot
// input stream uses non-temporal loads to keep weights L2-resident.
// ---------------------------------------------------------------------------

typedef __attribute__((ext_vector_type(16))) _Float16 v16h;
typedef __attribute__((ext_vector_type(8)))  _Float16 h8_t;
typedef __attribute__((ext_vector_type(4)))  _Float16 h4_t;
typedef __attribute__((ext_vector_type(8)))  float    v8f;
typedef __attribute__((ext_vector_type(4)))  float    f4v;

#define DEVFN static __device__ __forceinline__

// ---------------- packed-weight tile table (units: 32x16 tiles) -------------
#define TB_SW1   0      /* 256x128 : 8x8  = 64 */
#define TB_SW2   64     /* 128x96  : 4x6  = 24 */
#define TB_SW3   88     /* 96x64   : 3x4  = 12 */
#define TB_RB1W1 100    /* 64x64   : 2x4  = 8  */
#define TB_RB1W2 108
#define TB_RB2W1 116
#define TB_RB2W2 124
#define TB_EW1   132    /* 5 x (64x128 : 2x8 = 16) */
#define TB_EW2   212    /* 5 x (128x64 : 4x4 = 16) */
#define TB_EW3   292    /* 5 x (64x64  : 2x4 = 8 ) */
#define TB_GSW1  332    /* 64x48 -> 2x3 = 6 */
#define TB_GSW2  338    /* 48x5  -> 2x1 = 2 (padded) */
#define TB_GTW1  340    /* 3 x (64x32 : 2x2 = 4) */
#define TB_GTW2  352    /* 3 x (32x5 : 1x1 = 1, padded) */
#define TB_PSW   355    /* 64x64 : 8 */
#define TB_HW1   363    /* 3 x (64x32 : 4) */
#define TOTAL_TILES 375 /* * 512 halves = 192000 halves = 384000 B in d_ws */

struct Params {
  const float *x;
  const float *s_w1,*s_b1,*s_g1,*s_be1;
  const float *s_w2,*s_b2,*s_g2,*s_be2;
  const float *s_w3,*s_b3;
  const float *rb1_w1,*rb1_b1,*rb1_g1,*rb1_be1,*rb1_w2,*rb1_b2,*rb1_g2,*rb1_be2;
  const float *rb2_w1,*rb2_b1,*rb2_g1,*rb2_be1,*rb2_w2,*rb2_b2,*rb2_g2,*rb2_be2;
  const float *e_w1,*e_b1,*e_g1,*e_be1,*e_w2,*e_b2,*e_w3,*e_b3;
  const float *gs_w1,*gs_b1,*gs_w2,*gs_b2;
  const float *gst_w1,*gst_b1,*gst_w2,*gst_b2;
  const float *gte_w1,*gte_b1,*gte_w2,*gte_b2;
  const float *gyi_w1,*gyi_b1,*gyi_w2,*gyi_b2;
  const float *ps_w,*ps_b,*ps_g,*ps_be;
  const float *hst_w1,*hst_b1,*hst_w2,*hst_b2;
  const float *hte_w1,*hte_b1,*hte_w2,*hte_b2;
  const float *hyi_w1,*hyi_b1,*hyi_w2,*hyi_b2;
};

struct PackDesc { const float* src; int Ks, Ns, Kt, Nt, base; };
struct PackArgs { PackDesc m[34]; };

// --------------------------- weight packing kernel --------------------------
// Packed layout: tile*512 + lane*16 + i, where per lane: n = nt*16 + (lane&15),
// gsel = lane>>4, k = kt*32 + (i<8 ? gsel*8+i : 16+gsel*8+(i-8)).
// This mirrors the CDNA5 16-bit A/B fragment K striping (ISA 7.12.2).
__global__ void __launch_bounds__(256) pack_weights(PackArgs a, _Float16* out) {
  int gid = blockIdx.x * 256 + threadIdx.x;
  if (gid >= TOTAL_TILES * 512) return;
  int tile = gid >> 9;
  int idx  = gid & 511;
  int lane = idx >> 4;
  int i    = idx & 15;
  int mi = 0;
  for (int j = 0; j < 34; ++j) if (tile >= a.m[j].base) mi = j;
  PackDesc d = a.m[mi];
  int lt  = tile - d.base;
  int kt  = lt / d.Nt;
  int nt  = lt - kt * d.Nt;
  int col = lane & 15, gsel = lane >> 4;
  int n = nt * 16 + col;
  int k = kt * 32 + ((i < 8) ? (gsel * 8 + i) : (16 + gsel * 8 + (i - 8)));
  float v = 0.f;
  if (k < d.Ks && n < d.Ns) v = d.src[(size_t)k * d.Ns + n];
  out[gid] = (_Float16)v;
}

// ------------------------------ device helpers ------------------------------
DEVFN float silu_f(float x) { return x / (1.0f + __expf(-x)); }

DEVFN v8f vzero8() {
  v8f r;
#pragma unroll
  for (int i = 0; i < 8; ++i) r[i] = 0.f;
  return r;
}

template<int NT>
DEVFN void zero_acc(v8f* a) {
#pragma unroll
  for (int n = 0; n < NT; ++n) a[n] = vzero8();
}

// prefetch a run of packed weight tiles into the cache hierarchy
DEVFN void prefetch_tiles(const _Float16* p, int tiles, int lane) {
  const char* c = (const char*)p;
  const int bytes = tiles << 10;         // 1024 B per tile
  for (int o = lane * 128; o < bytes; o += 32 * 128)
    __builtin_prefetch(c + o, 0, 1);     // -> global_prefetch_b8
}

// A-fragment from LDS: 16x32 f16 tile, rows striped per-lane (ISA 7.12.2).
DEVFN v16h load_a_frag(const _Float16* act, int stride, int k0, int lane) {
  int m = lane & 15, gsel = lane >> 4;
  const _Float16* p = act + m * stride + k0 + gsel * 8;
  h8_t lo = *(const h8_t*)p;
  h8_t hi = *(const h8_t*)(p + 16);
  v16h r;
#pragma unroll
  for (int i = 0; i < 8; ++i) { r[i] = lo[i]; r[i + 8] = hi[i]; }
  return r;
}

// tiled GEMM: acc[NT] += act(16 x 32*KT) * W(32*KT x 16*NT).
// A fragments are preloaded; B fragments of each K-row land in independent
// registers so the global loads clause and overlap with WMMA issue.
template<int KT, int NT>
DEVFN void linear_acc(const _Float16* act, int stride, const _Float16* wt,
                      int lane, v8f* acc) {
  v16h afr[KT];
#pragma unroll
  for (int kt = 0; kt < KT; ++kt)
    afr[kt] = load_a_frag(act, stride, kt * 32, lane);
#pragma unroll
  for (int kt = 0; kt < KT; ++kt) {
    v16h b[NT];
#pragma unroll
    for (int nt = 0; nt < NT; ++nt)
      b[nt] = *(const v16h*)(wt + (size_t)((kt * NT + nt) << 9) + lane * 16);
#pragma unroll
    for (int nt = 0; nt < NT; ++nt)
      acc[nt] = __builtin_amdgcn_wmma_f32_16x16x32_f16(
          false, afr[kt], false, b[nt], (short)0, acc[nt], false, false);
  }
}

template<int NT>
DEVFN void bias_add(v8f* acc, const float* b, int lane) {
  int col = lane & 15;
#pragma unroll
  for (int nt = 0; nt < NT; ++nt) {
    float bn = b[nt * 16 + col];
#pragma unroll
    for (int v = 0; v < 8; ++v) acc[nt][v] += bn;
  }
}

// C-fragment scatter to f32 staging (stride fixed at 128 floats/row)
template<int NT>
DEVFN void scatter_f32(float* stage, const v8f* acc, int lane) {
  int col = lane & 15, gsel = lane >> 4;
#pragma unroll
  for (int nt = 0; nt < NT; ++nt)
#pragma unroll
    for (int v = 0; v < 8; ++v)
      stage[(gsel * 8 + v) * 128 + nt * 16 + col] = acc[nt][v];
}

// C-fragment scatter to f16 activation buffer (optional SiLU)
template<int NT>
DEVFN void scatter_f16(_Float16* out, int stride, const v8f* acc, int lane,
                       bool doSilu) {
  int col = lane & 15, gsel = lane >> 4;
#pragma unroll
  for (int nt = 0; nt < NT; ++nt)
#pragma unroll
    for (int v = 0; v < 8; ++v) {
      float x = acc[nt][v];
      if (doSilu) x = silu_f(x);
      out[(gsel * 8 + v) * stride + nt * 16 + col] = (_Float16)x;
    }
}

// row-wise LayerNorm (+ optional SiLU / residual fuse), lanes 0..15 = rows.
// stage stride = 128 f32.  resid != null: h = silu(h + LN(...)) into resid
// (stride 64 f16).  Otherwise writes f16 rows to outAct with outStride.
DEVFN void ln_rows(const float* stage, int W, const float* g, const float* be,
                   _Float16* outAct, int outStride, int doSilu,
                   _Float16* resid, int lane) {
  if (lane >= 16) return;
  const float* row = stage + lane * 128;
  float s = 0.f;
  for (int k = 0; k < W; k += 4) {
    f4v v = *(const f4v*)(row + k);
    s += v[0] + v[1] + v[2] + v[3];
  }
  float mean = s / (float)W;
  float q = 0.f;
  for (int k = 0; k < W; k += 4) {
    f4v v = *(const f4v*)(row + k);
#pragma unroll
    for (int j = 0; j < 4; ++j) { float d = v[j] - mean; q += d * d; }
  }
  float inv = rsqrtf(q / (float)W + 1e-5f);
  for (int k = 0; k < W; k += 4) {
    f4v v  = *(const f4v*)(row + k);
    f4v gg = *(const f4v*)(g + k);
    f4v bb = *(const f4v*)(be + k);
    float y[4];
#pragma unroll
    for (int j = 0; j < 4; ++j) y[j] = (v[j] - mean) * inv * gg[j] + bb[j];
    if (resid) {
#pragma unroll
      for (int j = 0; j < 4; ++j) {
        float z = (float)resid[lane * 64 + k + j] + y[j];
        resid[lane * 64 + k + j] = (_Float16)silu_f(z);
      }
    } else {
      h4_t h;
#pragma unroll
      for (int j = 0; j < 4; ++j) {
        float z = doSilu ? silu_f(y[j]) : y[j];
        h[j] = (_Float16)z;
      }
      *(h4_t*)(outAct + lane * outStride + k) = h;
    }
  }
}

// ------------------------------- main kernel --------------------------------
#define WAVES 2  // 2 waves/block * 16 rows = 32 rows per block

__global__ void __launch_bounds__(64) moe_fwd(Params P, const _Float16* Wt,
                                              float* out) {
  __shared__ _Float16 sAct[WAVES][16 * 256];   // f16 activations (8KB/wave)
  __shared__ float    sStage[WAVES][16 * 128]; // f32 LN staging (8KB/wave)
  __shared__ _Float16 sH[WAVES][16 * 64];      // trunk feature h (2KB/wave)
  __shared__ float    sLog[WAVES][3 * 16 * 8]; // logits -> gate weights

  const int lane = threadIdx.x & 31;
  const int wid  = threadIdx.x >> 5;
  _Float16* act   = sAct[wid];
  float*    stage = sStage[wid];
  _Float16* hbuf  = sH[wid];
  float*    logb  = sLog[wid];
  const int rowBase = (blockIdx.x * WAVES + wid) * 16;
  const int col = lane & 15, gsel = lane >> 4;

  // warm L2/L1 for the first stages while we stream the input tile
  prefetch_tiles(Wt + (size_t)TB_SW1 * 512, 64, lane);

  // ---- load x tile (16x256 f32) -> f16 LDS, non-temporal (streamed once) ----
  {
    const float* xp = P.x + (size_t)rowBase * 256;
#pragma unroll
    for (int j = 0; j < 32; ++j) {
      int off = (j * 32 + lane) * 4;
      f4v v = __builtin_nontemporal_load((const f4v*)(xp + off));
      h4_t h;
#pragma unroll
      for (int q = 0; q < 4; ++q) h[q] = (_Float16)v[q];
      *(h4_t*)(act + off) = h;
    }
  }

  v8f acc[8];

  // ---- stem 1: 256 -> 128, LN, SiLU ----
  prefetch_tiles(Wt + (size_t)TB_SW2 * 512, 36, lane);  // stem2+stem3
  zero_acc<8>(acc);
  linear_acc<8, 8>(act, 256, Wt + (size_t)TB_SW1 * 512, lane, acc);
  bias_add<8>(acc, P.s_b1, lane);
  scatter_f32<8>(stage, acc, lane);
  ln_rows(stage, 128, P.s_g1, P.s_be1, act, 128, 1, nullptr, lane);

  // ---- stem 2: 128 -> 96, LN, SiLU ----
  prefetch_tiles(Wt + (size_t)TB_RB1W1 * 512, 32, lane);  // trunk
  zero_acc<6>(acc);
  linear_acc<4, 6>(act, 128, Wt + (size_t)TB_SW2 * 512, lane, acc);
  bias_add<6>(acc, P.s_b2, lane);
  scatter_f32<6>(stage, acc, lane);
  ln_rows(stage, 96, P.s_g2, P.s_be2, act, 96, 1, nullptr, lane);

  // ---- stem 3: 96 -> 64, SiLU -> h ----
  prefetch_tiles(Wt + (size_t)TB_GSW1 * 512, 23, lane);  // all gate weights
  zero_acc<4>(acc);
  linear_acc<3, 4>(act, 96, Wt + (size_t)TB_SW3 * 512, lane, acc);
  bias_add<4>(acc, P.s_b3, lane);
  scatter_f16<4>(hbuf, 64, acc, lane, true);

  // ---- trunk: 2 residual blocks ----
  {
    const _Float16* w1t[2] = { Wt + (size_t)TB_RB1W1 * 512, Wt + (size_t)TB_RB2W1 * 512 };
    const _Float16* w2t[2] = { Wt + (size_t)TB_RB1W2 * 512, Wt + (size_t)TB_RB2W2 * 512 };
    const float* b1[2]  = { P.rb1_b1,  P.rb2_b1  };
    const float* g1[2]  = { P.rb1_g1,  P.rb2_g1  };
    const float* be1[2] = { P.rb1_be1, P.rb2_be1 };
    const float* b2[2]  = { P.rb1_b2,  P.rb2_b2  };
    const float* g2[2]  = { P.rb1_g2,  P.rb2_g2  };
    const float* be2[2] = { P.rb1_be2, P.rb2_be2 };
    for (int i = 0; i < 2; ++i) {
      zero_acc<4>(acc);
      linear_acc<2, 4>(hbuf, 64, w1t[i], lane, acc);
      bias_add<4>(acc, b1[i], lane);
      scatter_f32<4>(stage, acc, lane);
      ln_rows(stage, 64, g1[i], be1[i], act, 64, 1, nullptr, lane);
      zero_acc<4>(acc);
      linear_acc<2, 4>(act, 64, w2t[i], lane, acc);
      bias_add<4>(acc, b2[i], lane);
      scatter_f32<4>(stage, acc, lane);
      ln_rows(stage, 64, g2[i], be2[i], nullptr, 0, 0, hbuf, lane); // h=silu(h+t)
    }
  }

  // ---- shared gate: 64 -> 48 (SiLU) -> 5, logits -> stage[16x8] ----
  prefetch_tiles(Wt + (size_t)TB_EW1 * 512, 16, lane);  // expert 0 w1
  zero_acc<3>(acc);
  linear_acc<2, 3>(hbuf, 64, Wt + (size_t)TB_GSW1 * 512, lane, acc);
  bias_add<3>(acc, P.gs_b1, lane);
  scatter_f16<3>(act, 64, acc, lane, true);
#pragma unroll
  for (int v = 0; v < 8; ++v)  // zero-pad cols 48..63 (K padded to 64)
    act[(gsel * 8 + v) * 64 + 48 + col] = (_Float16)0.f;
  zero_acc<1>(acc);
  linear_acc<2, 1>(act, 64, Wt + (size_t)TB_GSW2 * 512, lane, acc);
  if (col < 5) {
    float b = P.gs_b2[col];
#pragma unroll
    for (int v = 0; v < 8; ++v)
      stage[(gsel * 8 + v) * 8 + col] = acc[0][v] + b;
  }

  // ---- task gates: logits[t] = shared + task(h) ----
  {
    const _Float16* gw1t[3] = { Wt + (size_t)(TB_GTW1 + 0) * 512,
                                Wt + (size_t)(TB_GTW1 + 4) * 512,
                                Wt + (size_t)(TB_GTW1 + 8) * 512 };
    const _Float16* gw2t[3] = { Wt + (size_t)(TB_GTW2 + 0) * 512,
                                Wt + (size_t)(TB_GTW2 + 1) * 512,
                                Wt + (size_t)(TB_GTW2 + 2) * 512 };
    const float* gb1[3] = { P.gst_b1, P.gte_b1, P.gyi_b1 };
    const float* gb2[3] = { P.gst_b2, P.gte_b2, P.gyi_b2 };
    for (int t = 0; t < 3; ++t) {
      zero_acc<2>(acc);
      linear_acc<2, 2>(hbuf, 64, gw1t[t], lane, acc);
      bias_add<2>(acc, gb1[t], lane);
      scatter_f16<2>(act, 32, acc, lane, true);
      zero_acc<1>(acc);
      linear_acc<1, 1>(act, 32, gw2t[t], lane, acc);
      if (col < 5) {
        float b = gb2[t][col];
#pragma unroll
        for (int v = 0; v < 8; ++v) {
          int m = gsel * 8 + v;
          logb[t * 128 + m * 8 + col] = acc[0][v] + b + stage[m * 8 + col];
        }
      }
    }
  }

  // ---- softmax over E=5 per (task,row), in place -> gate weights ----
  if (lane < 16) {
    for (int t = 0; t < 3; ++t) {
      float* lr = logb + t * 128 + lane * 8;
      float mx = lr[0];
      for (int e = 1; e < 5; ++e) mx = fmaxf(mx, lr[e]);
      float s = 0.f, ex[5];
      for (int e = 0; e < 5; ++e) { ex[e] = __expf(lr[e] - mx); s += ex[e]; }
      float inv = 1.f / s;
      for (int e = 0; e < 5; ++e) lr[e] = ex[e] * inv;
    }
  }

  // ---- experts: accumulate gated outputs into feat[t] (C-layout regs) ----
  v8f feat[3][4];
#pragma unroll
  for (int t = 0; t < 3; ++t) zero_acc<4>(feat[t]);

  for (int e = 0; e < 5; ++e) {
    const _Float16* w1 = Wt + (size_t)(TB_EW1 + e * 16) * 512;
    const _Float16* w2 = Wt + (size_t)(TB_EW2 + e * 16) * 512;
    const _Float16* w3 = Wt + (size_t)(TB_EW3 + e * 8)  * 512;
    // prefetch the next expert (or the post/head weights on the last one)
    if (e < 4) {
      prefetch_tiles(Wt + (size_t)(TB_EW1 + (e + 1) * 16) * 512, 16, lane);
      prefetch_tiles(Wt + (size_t)(TB_EW2 + (e + 1) * 16) * 512, 16, lane);
      prefetch_tiles(Wt + (size_t)(TB_EW3 + (e + 1) * 8)  * 512,  8, lane);
    } else {
      prefetch_tiles(Wt + (size_t)TB_PSW * 512, 20, lane);  // ps_w + head w1s
    }
    // e1 = silu(LN(h @ w1 + b1))
    zero_acc<8>(acc);
    linear_acc<2, 8>(hbuf, 64, w1, lane, acc);
    bias_add<8>(acc, P.e_b1 + e * 128, lane);
    scatter_f32<8>(stage, acc, lane);
    ln_rows(stage, 128, P.e_g1 + e * 128, P.e_be1 + e * 128, act, 128, 1,
            nullptr, lane);
    // e2 = silu(e1 @ w2 + b2)
    zero_acc<4>(acc);
    linear_acc<4, 4>(act, 128, w2, lane, acc);
    bias_add<4>(acc, P.e_b2 + e * 64, lane);
    scatter_f16<4>(act, 64, acc, lane, true);
    // eo = silu(e2 @ w3 + b3); feat[t] += gw[t][row][e] * eo
    zero_acc<4>(acc);
    linear_acc<2, 4>(act, 64, w3, lane, acc);
    bias_add<4>(acc, P.e_b3 + e * 64, lane);
#pragma unroll
    for (int v = 0; v < 8; ++v) {
      int m = gsel * 8 + v;
      float g0 = logb[0 * 128 + m * 8 + e];
      float g1 = logb[1 * 128 + m * 8 + e];
      float g2 = logb[2 * 128 + m * 8 + e];
#pragma unroll
      for (int nt = 0; nt < 4; ++nt) {
        float so = silu_f(acc[nt][v]);
        feat[0][nt][v] += so * g0;
        feat[1][nt][v] += so * g1;
        feat[2][nt][v] += so * g2;
      }
    }
  }

  // ---- per-task post_shared + head ----
  {
    const _Float16* hw1t[3] = { Wt + (size_t)(TB_HW1 + 0) * 512,
                                Wt + (size_t)(TB_HW1 + 4) * 512,
                                Wt + (size_t)(TB_HW1 + 8) * 512 };
    const float* hb1[3] = { P.hst_b1, P.hte_b1, P.hyi_b1 };
    const float* hw2[3] = { P.hst_w2, P.hte_w2, P.hyi_w2 };
    const float* hb2[3] = { P.hst_b2, P.hte_b2, P.hyi_b2 };
    for (int t = 0; t < 3; ++t) {
      scatter_f16<4>(act, 64, feat[t], lane, false);    // feat -> f16 LDS
      zero_acc<4>(acc);
      linear_acc<2, 4>(act, 64, Wt + (size_t)TB_PSW * 512, lane, acc);
      bias_add<4>(acc, P.ps_b, lane);
      scatter_f32<4>(stage, acc, lane);
      ln_rows(stage, 64, P.ps_g, P.ps_be, act, 64, 1, nullptr, lane);
      zero_acc<2>(acc);
      linear_acc<2, 2>(act, 64, hw1t[t], lane, acc);
      bias_add<2>(acc, hb1[t], lane);
      scatter_f16<2>(act, 32, acc, lane, true);
      if (lane < 16) {   // head2: 32 -> 1 scalar dot
        float s = hb2[t][0];
        for (int k = 0; k < 32; ++k)
          s += (float)act[lane * 32 + k] * hw2[t][k];
        out[(size_t)(rowBase + lane) * 3 + t] = s;
      }
    }
  }
}

// ------------------------------- host launch --------------------------------
extern "C" void kernel_launch(void* const* d_in, const int* in_sizes, int n_in,
                              void* d_out, int out_size, void* d_ws,
                              size_t ws_size, hipStream_t stream) {
  (void)in_sizes; (void)n_in; (void)out_size; (void)ws_size;
  auto F = [&](int i) { return (const float*)d_in[i]; };

  Params P;
  P.x = F(0);
  P.s_w1 = F(1);  P.s_b1 = F(2);  P.s_g1 = F(3);  P.s_be1 = F(4);
  P.s_w2 = F(5);  P.s_b2 = F(6);  P.s_g2 = F(7);  P.s_be2 = F(8);
  P.s_w3 = F(9);  P.s_b3 = F(10);
  P.rb1_w1 = F(11); P.rb1_b1 = F(12); P.rb1_g1 = F(13); P.rb1_be1 = F(14);
  P.rb1_w2 = F(15); P.rb1_b2 = F(16); P.rb1_g2 = F(17); P.rb1_be2 = F(18);
  P.rb2_w1 = F(19); P.rb2_b1 = F(20); P.rb2_g1 = F(21); P.rb2_be1 = F(22);
  P.rb2_w2 = F(23); P.rb2_b2 = F(24); P.rb2_g2 = F(25); P.rb2_be2 = F(26);
  P.e_w1 = F(27); P.e_b1 = F(28); P.e_g1 = F(29); P.e_be1 = F(30);
  P.e_w2 = F(31); P.e_b2 = F(32); P.e_w3 = F(33); P.e_b3 = F(34);
  P.gs_w1 = F(35); P.gs_b1 = F(36); P.gs_w2 = F(37); P.gs_b2 = F(38);
  P.gst_w1 = F(39); P.gst_b1 = F(40); P.gst_w2 = F(41); P.gst_b2 = F(42);
  P.gte_w1 = F(43); P.gte_b1 = F(44); P.gte_w2 = F(45); P.gte_b2 = F(46);
  P.gyi_w1 = F(47); P.gyi_b1 = F(48); P.gyi_w2 = F(49); P.gyi_b2 = F(50);
  P.ps_w = F(51); P.ps_b = F(52); P.ps_g = F(53); P.ps_be = F(54);
  P.hst_w1 = F(55); P.hst_b1 = F(56); P.hst_w2 = F(57); P.hst_b2 = F(58);
  P.hte_w1 = F(59); P.hte_b1 = F(60); P.hte_w2 = F(61); P.hte_b2 = F(62);
  P.hyi_w1 = F(63); P.hyi_b1 = F(64); P.hyi_w2 = F(65); P.hyi_b2 = F(66);

  PackArgs pa;
  int d = 0;
  auto add = [&](const float* s, int Ks, int Ns, int Kt, int Nt, int base) {
    pa.m[d].src = s; pa.m[d].Ks = Ks; pa.m[d].Ns = Ns;
    pa.m[d].Kt = Kt; pa.m[d].Nt = Nt; pa.m[d].base = base; ++d;
  };
  add(P.s_w1, 256, 128, 8, 8, TB_SW1);
  add(P.s_w2, 128, 96, 4, 6, TB_SW2);
  add(P.s_w3, 96, 64, 3, 4, TB_SW3);
  add(P.rb1_w1, 64, 64, 2, 4, TB_RB1W1);
  add(P.rb1_w2, 64, 64, 2, 4, TB_RB1W2);
  add(P.rb2_w1, 64, 64, 2, 4, TB_RB2W1);
  add(P.rb2_w2, 64, 64, 2, 4, TB_RB2W2);
  for (int e = 0; e < 5; ++e) add(P.e_w1 + e * 64 * 128, 64, 128, 2, 8, TB_EW1 + e * 16);
  for (int e = 0; e < 5; ++e) add(P.e_w2 + e * 128 * 64, 128, 64, 4, 4, TB_EW2 + e * 16);
  for (int e = 0; e < 5; ++e) add(P.e_w3 + e * 64 * 64, 64, 64, 2, 4, TB_EW3 + e * 8);
  add(P.gs_w1, 64, 48, 2, 3, TB_GSW1);
  add(P.gs_w2, 48, 5, 2, 1, TB_GSW2);
  add(P.gst_w1, 64, 32, 2, 2, TB_GTW1 + 0);
  add(P.gte_w1, 64, 32, 2, 2, TB_GTW1 + 4);
  add(P.gyi_w1, 64, 32, 2, 2, TB_GTW1 + 8);
  add(P.gst_w2, 32, 5, 1, 1, TB_GTW2 + 0);
  add(P.gte_w2, 32, 5, 1, 1, TB_GTW2 + 1);
  add(P.gyi_w2, 32, 5, 1, 1, TB_GTW2 + 2);
  add(P.ps_w, 64, 64, 2, 4, TB_PSW);
  add(P.hst_w1, 64, 32, 2, 2, TB_HW1 + 0);
  add(P.hte_w1, 64, 32, 2, 2, TB_HW1 + 4);
  add(P.hyi_w1, 64, 32, 2, 2, TB_HW1 + 8);

  _Float16* wpack = (_Float16*)d_ws;
  const int packElems = TOTAL_TILES * 512;
  pack_weights<<<(packElems + 255) / 256, 256, 0, stream>>>(pa, wpack);

  const int B = 131072;
  const int blocks = B / (WAVES * 16);  // 4096
  moe_fwd<<<blocks, WAVES * 32, 0, stream>>>(P, (const _Float16*)wpack,
                                             (float*)d_out);
}
